// Attention_65326452572738
// MI455X (gfx1250) — matrix operand order repaired and checked by reading, not compiled, not run on hardware
//
#include <hip/hip_runtime.h>

// ---------------------------------------------------------------------------
// Problem constants (from reference)
// ---------------------------------------------------------------------------
#define BATCH   4
#define CHANS   256
#define IMH     128
#define IMW     128
#define NTOK    (IMH * IMW)          // 16384 tokens per image
#define BN      (BATCH * NTOK)       // 65536 total tokens
#define NHEADS  8
#define HD      32                   // head dim
#define KCONV   (9 * CHANS)          // 2304 implicit-GEMM K

typedef __bf16 bf16_t;
typedef __attribute__((ext_vector_type(16))) __bf16 v16bf;
typedef __attribute__((ext_vector_type(8)))  float  v8f;
typedef __attribute__((ext_vector_type(4)))  unsigned uint4v;

union V16U { v16bf v; uint4v q[2]; unsigned u[8]; };

// ---------------------------------------------------------------------------
// bf16 helpers (bit-level, no __bf16 arithmetic needed)
// ---------------------------------------------------------------------------
static __device__ inline bf16_t f2bf(float f) {
    unsigned u = __builtin_bit_cast(unsigned, f);
    unsigned r = (u + 0x7FFFu + ((u >> 16) & 1u)) >> 16;   // RNE
    unsigned short s = (unsigned short)r;
    return __builtin_bit_cast(bf16_t, s);
}
static __device__ inline float bf2f(bf16_t b) {
    unsigned short s = __builtin_bit_cast(unsigned short, b);
    unsigned u = ((unsigned)s) << 16;
    return __builtin_bit_cast(float, u);
}
static __device__ inline unsigned pack2(bf16_t lo, bf16_t hi) {
    return (unsigned)__builtin_bit_cast(unsigned short, lo) |
           ((unsigned)__builtin_bit_cast(unsigned short, hi) << 16);
}

// LDS byte address (wave-relative) of a generic pointer to __shared__ data:
// per CDNA5 ISA §10.2 the low 32 bits of a generic LDS address are the
// wave-relative LDS offset, which is exactly what async-to-LDS VDST wants.
static __device__ inline unsigned lds_addr_of(const void* p) {
    return (unsigned)(unsigned long long)p;
}

// Async global -> LDS copy of 32 bytes (2 x b128), tracked by ASYNCcnt.
static __device__ inline void async_copy32(unsigned lds_b, const void* gptr) {
    asm volatile(
        "global_load_async_to_lds_b128 %0, %1, off\n\t"
        "global_load_async_to_lds_b128 %0, %1, off offset:16"
        :: "v"(lds_b), "v"(gptr) : "memory");
}
static __device__ inline void wait_async0() {
    asm volatile("s_wait_asynccnt 0" ::: "memory");
}

// ---------------------------------------------------------------------------
// Fragment loaders matching CDNA5 WMMA bf16 VGPR layouts (ISA 7.12.2).
// A 16x32: lanes 0-15 -> M=lane, K-halves {kb..kb+7, 16+kb..16+kb+7}, kb=(lane<16)?0:8
// B 32x16: lanes 0-15 -> N=lane, K-halves {kb..kb+7, 16+kb..16+kb+7}, kb=(lane<16)?0:8
// ---------------------------------------------------------------------------
static __device__ inline v16bf ldA16(const bf16_t* row, int lane) {
    V16U t; int kb = (lane & 16) ? 8 : 0;
    t.q[0] = *(const uint4v*)(row + kb);
    t.q[1] = *(const uint4v*)(row + 16 + kb);
    return t.v;
}
static __device__ inline v16bf ldB16(const bf16_t* row, int lane) {
    V16U t; int kb = (lane & 16) ? 8 : 0;
    t.q[0] = *(const uint4v*)(row + kb);
    t.q[1] = *(const uint4v*)(row + 16 + kb);
    return t.v;
}

static __device__ inline float wave_sum(float x) {
    #pragma unroll
    for (int o = 16; o > 0; o >>= 1) x += __shfl_xor(x, o, 32);
    return x;
}

// ---------------------------------------------------------------------------
// 1) Weight repack: w[Cout][Cin][3][3] f32 -> bf16 [Cout][K], K = t*256 + ci
// ---------------------------------------------------------------------------
__global__ __launch_bounds__(256) void prep_w_kernel(const float* __restrict__ w,
                                                     bf16_t* __restrict__ out) {
    int co = blockIdx.x, ci = threadIdx.x;
    #pragma unroll
    for (int t = 0; t < 9; ++t)
        out[(size_t)co * KCONV + t * CHANS + ci] =
            f2bf(w[((size_t)(co * CHANS + ci)) * 9 + t]);
}

// ---------------------------------------------------------------------------
// 2) NCHW f32 -> token-major bf16 [B][N][C]
// ---------------------------------------------------------------------------
__global__ __launch_bounds__(256) void tok_bf16_kernel(const float* __restrict__ src,
                                                       bf16_t* __restrict__ dst) {
    int bc = blockIdx.x;                // b*256 + c
    int b = bc >> 8, c = bc & 255;
    const float* s = src + (size_t)bc * NTOK;
    for (int n = threadIdx.x; n < NTOK; n += 256)
        dst[((size_t)(b << 14) + n) * CHANS + c] = f2bf(s[n]);
}

// ---------------------------------------------------------------------------
// 3) Implicit-GEMM 3x3 conv via v_wmma_f32_16x16x32_bf16.
//    Block: 128 threads (4 waves). Tile: 64 tokens x 64 out-channels.
//    Double-buffered LDS tiles filled with GLOBAL_LOAD_ASYNC_TO_LDS_B128
//    (ASYNCcnt pipeline): issue stage(k+1) async, run WMMAs on stage(k),
//    s_wait_asynccnt 0, barrier, swap.
//    out_nchw==0 -> f32 token-major [BN][256]; ==1 -> f32 NCHW [B][256][H][W]
// ---------------------------------------------------------------------------
__global__ __launch_bounds__(128) void conv3x3_wmma_kernel(
    const bf16_t* __restrict__ inTok, const bf16_t* __restrict__ wprep,
    const float* __restrict__ bias, float* __restrict__ out, int out_nchw)
{
    __shared__ __align__(16) bf16_t As[2][64][32];   // [buf][token][ci-chunk]
    __shared__ __align__(16) bf16_t Bs[2][64][32];   // [buf][outch][k-chunk]

    const int tid = threadIdx.x, lane = tid & 31, wvid = tid >> 5;
    const int lh = lane & 15;
    const int m0 = blockIdx.x * 64;        // token tile base (same batch for all 64)
    const int col0 = blockIdx.y * 64;      // out-channel tile base
    const int b = m0 >> 14;

    v8f acc[4] = {};

    // staging roles: thread -> (row 0..63, 16-half chunk 0/16)
    const int a_tok  = tid >> 1;
    const int a_half = (tid & 1) * 16;
    const int tokn = (m0 & (NTOK - 1)) + a_tok;
    const int th = tokn >> 7, tw = tokn & 127;
    const bf16_t* wrow = wprep + (size_t)(col0 + a_tok) * KCONV + a_half;

    // stage K-step kk (= t*8 + ci-chunk) into LDS buffer `buf`
    auto stage = [&](int kk, int buf) {
        const int t = kk >> 3, c0 = (kk & 7) * 32;
        const int dh = t / 3 - 1, dw = t % 3 - 1;
        const int hh = th + dh, ww = tw + dw;
        const bool valid = (hh >= 0) && (hh < IMH) && (ww >= 0) && (ww < IMW);
        if (valid) {
            const bf16_t* ga = inTok +
                ((size_t)(b * NTOK + hh * IMW + ww)) * CHANS + c0 + a_half;
            async_copy32(lds_addr_of(&As[buf][a_tok][a_half]), ga);
        } else {
            *(uint4v*)&As[buf][a_tok][a_half]     = (uint4v)0u;
            *(uint4v*)&As[buf][a_tok][a_half + 8] = (uint4v)0u;
        }
        const bf16_t* gb = wrow + t * CHANS + c0;
        async_copy32(lds_addr_of(&Bs[buf][a_tok][a_half]), gb);
    };

    stage(0, 0);
    wait_async0();
    __syncthreads();

    for (int kk = 0; kk < 72; ++kk) {
        const int buf = kk & 1;
        if (kk < 71) stage(kk + 1, buf ^ 1);     // prefetch next K-step (async)

        const v16bf bfrag = ldB16(&Bs[buf][wvid * 16 + lh][0], lane);
        #pragma unroll
        for (int mf = 0; mf < 4; ++mf) {
            const v16bf afrag = ldA16(&As[buf][mf * 16 + lh][0], lane);
            acc[mf] = __builtin_amdgcn_wmma_f32_16x16x32_bf16(
                false, afrag, false, bfrag, (short)0, acc[mf], false, false);
        }

        wait_async0();        // our wave's async writes to buf^1 are in LDS
        __syncthreads();      // (compiler adds dscnt wait for the zero-fill path)
    }

    // epilogue: C/D layout -> lane = N, VGPR r = M row (+8 for upper half-wave)
    const int co = col0 + wvid * 16 + lh;
    const float bsv = bias[co];
    #pragma unroll
    for (int mf = 0; mf < 4; ++mf) {
        #pragma unroll
        for (int r = 0; r < 8; ++r) {
            const int m = m0 + mf * 16 + r + ((lane & 16) ? 8 : 0);
            const float val = acc[mf][r] + bsv;
            if (out_nchw) {
                const int bb = m >> 14, nn = m & (NTOK - 1);
                out[((size_t)(bb * CHANS + co) << 14) + nn] = val;
            } else {
                out[(size_t)m * CHANS + co] = val;
            }
        }
    }
}

// ---------------------------------------------------------------------------
// 4) zero helper
// ---------------------------------------------------------------------------
__global__ __launch_bounds__(256) void zero_kernel(float* p, int n) {
    int i = blockIdx.x * 256 + threadIdx.x;
    if (i < n) p[i] = 0.f;
}

// ---------------------------------------------------------------------------
// 5) focus + normalize. One wave per token; lane owns 8 channels.
//    q_final = q^6 / ||q^6|| (leaky 0.1 first); same for k; v_final = v/||v||.
//    Also accumulates ksum[b][c] = sum_n k_final.
// ---------------------------------------------------------------------------
__global__ __launch_bounds__(256) void normalize_kernel(
    const float* __restrict__ qf, const float* __restrict__ kf, const float* __restrict__ vf,
    bf16_t* __restrict__ qbf, bf16_t* __restrict__ kbf, bf16_t* __restrict__ vbf,
    float* __restrict__ ksum)
{
    __shared__ float ksacc[CHANS];
    const int tid = threadIdx.x, lane = tid & 31, wvid = tid >> 5;
    ksacc[tid] = 0.f;
    __syncthreads();

    const int gm = blockIdx.x * 8 + wvid;   // global token
    const int b = gm >> 14;
    const size_t base = (size_t)gm * CHANS + lane * 8;

    float t6[8], ss, inv;

    // q
    ss = 0.f;
    #pragma unroll
    for (int i = 0; i < 8; ++i) {
        float t = qf[base + i]; t = t > 0.f ? t : 0.1f * t;
        float t2 = t * t; float v6 = t2 * t2 * t2;
        t6[i] = v6; ss += v6 * v6;
    }
    inv = rsqrtf(fmaxf(wave_sum(ss), 1e-24f));
    #pragma unroll
    for (int i = 0; i < 8; ++i) qbf[base + i] = f2bf(t6[i] * inv);

    // k (+ ksum)
    ss = 0.f;
    #pragma unroll
    for (int i = 0; i < 8; ++i) {
        float t = kf[base + i]; t = t > 0.f ? t : 0.1f * t;
        float t2 = t * t; float v6 = t2 * t2 * t2;
        t6[i] = v6; ss += v6 * v6;
    }
    inv = rsqrtf(fmaxf(wave_sum(ss), 1e-24f));
    #pragma unroll
    for (int i = 0; i < 8; ++i) {
        float kn = t6[i] * inv;
        kbf[base + i] = f2bf(kn);
        atomicAdd(&ksacc[lane * 8 + i], kn);
    }

    // v
    ss = 0.f;
    #pragma unroll
    for (int i = 0; i < 8; ++i) { float t = vf[base + i]; t6[i] = t; ss += t * t; }
    inv = rsqrtf(fmaxf(wave_sum(ss), 1e-24f));
    #pragma unroll
    for (int i = 0; i < 8; ++i) vbf[base + i] = f2bf(t6[i] * inv);

    __syncthreads();
    atomicAdd(&ksum[(size_t)b * CHANS + tid], ksacc[tid]);
}

// ---------------------------------------------------------------------------
// 6) z[bh][n] = 1 / (q[n] . ksum_head + eps)
// ---------------------------------------------------------------------------
__global__ __launch_bounds__(256) void z_kernel(const bf16_t* __restrict__ qbf,
                                                const float* __restrict__ ksum,
                                                float* __restrict__ zb)
{
    const int g = blockIdx.x * 256 + threadIdx.x;  // (token, head)
    const int gm = g >> 3, head = g & 7;
    const int b = gm >> 14, n = gm & (NTOK - 1);
    const bf16_t* qrow = qbf + (size_t)gm * CHANS + head * HD;
    const float* ks = ksum + (size_t)b * CHANS + head * HD;
    float dot = 0.f;
    #pragma unroll
    for (int c = 0; c < HD; ++c) dot += bf2f(qrow[c]) * ks[c];
    zb[(size_t)(b * NHEADS + head) * NTOK + n] = 1.0f / (dot + 1e-6f);
}

// ---------------------------------------------------------------------------
// 7) kv = k^T v per (b,head): [32x32] over K=16384, WMMA with LDS transpose.
//    One block per bh, 4 waves split K, reduce partials in LDS.
//    Stores kv TRANSPOSED as kvbf[bh][d][c] so stage-9 B-frags are row-contiguous.
// ---------------------------------------------------------------------------
__global__ __launch_bounds__(128) void kv_kernel(const bf16_t* __restrict__ kbf,
                                                 const bf16_t* __restrict__ vbf,
                                                 bf16_t* __restrict__ kvbf)
{
    __shared__ __align__(16) bf16_t ks[4][32][32];
    __shared__ __align__(16) bf16_t vs[4][32][32];
    __shared__ float red[4][32][32];

    const int bh = blockIdx.x, b = bh >> 3, head = bh & 7;
    const int tid = threadIdx.x, lane = tid & 31, wvid = tid >> 5;
    const int lh = lane & 15;
    const int kbA = (lane & 16) ? 8 : 0;
    const int kbB = (lane & 16) ? 16 : 0;

    v8f acc[2][2] = {};

    for (int n0 = wvid * 32; n0 < NTOK; n0 += 128) {
        // stage: lane loads token (n0+lane)'s 32-channel head slice of k and v
        const size_t toff = ((size_t)(b * NTOK + n0 + lane)) * CHANS + head * HD;
        const uint4v* kp = (const uint4v*)(kbf + toff);
        const uint4v* vp = (const uint4v*)(vbf + toff);
        uint4v* kd = (uint4v*)&ks[wvid][lane][0];
        uint4v* vd = (uint4v*)&vs[wvid][lane][0];
        #pragma unroll
        for (int i = 0; i < 4; ++i) { kd[i] = kp[i]; vd[i] = vp[i]; }
        __syncthreads();

        V16U bf0, bf1;
        #pragma unroll
        for (int r = 0; r < 8; ++r) {
            int kk = (r < 4) ? (kbA + 2 * r) : (16 + kbA + 2 * (r - 4));
            bf0.u[r] = pack2(vs[wvid][kk][lh],      vs[wvid][kk + 1][lh]);
            bf1.u[r] = pack2(vs[wvid][kk][16 + lh], vs[wvid][kk + 1][16 + lh]);
        }
        #pragma unroll
        for (int mf = 0; mf < 2; ++mf) {
            const int m = mf * 16 + lh;   // c row of k^T
            V16U a;
            #pragma unroll
            for (int r = 0; r < 8; ++r) {
                int kk = (r < 4) ? (kbA + 2 * r) : (16 + kbA + 2 * (r - 4));
                a.u[r] = pack2(ks[wvid][kk][m], ks[wvid][kk + 1][m]);
            }
            acc[mf][0] = __builtin_amdgcn_wmma_f32_16x16x32_bf16(
                false, a.v, false, bf0.v, (short)0, acc[mf][0], false, false);
            acc[mf][1] = __builtin_amdgcn_wmma_f32_16x16x32_bf16(
                false, a.v, false, bf1.v, (short)0, acc[mf][1], false, false);
        }
        __syncthreads();
    }

    // per-wave partial kv -> LDS
    #pragma unroll
    for (int mf = 0; mf < 2; ++mf)
        #pragma unroll
        for (int nf = 0; nf < 2; ++nf)
            #pragma unroll
            for (int r = 0; r < 8; ++r) {
                const int m = mf * 16 + r + ((lane & 16) ? 8 : 0);
                const int d = nf * 16 + lh;
                red[wvid][m][d] = acc[mf][nf][r];
            }
    __syncthreads();
    for (int e = tid; e < 32 * 32; e += 128) {
        const int m = e >> 5, d = e & 31;   // m = c (of k), d (of v)
        const float s = red[0][m][d] + red[1][m][d] + red[2][m][d] + red[3][m][d];
        kvbf[(size_t)bh * (HD * HD) + d * HD + m] = f2bf(s);  // stored [d][c]
    }
}

// ---------------------------------------------------------------------------
// 8) depthwise 3x3 on v (per head image) -> fm, token-major bf16 [B][N][256]
// ---------------------------------------------------------------------------
__global__ __launch_bounds__(256) void fm_kernel(const bf16_t* __restrict__ vbf,
                                                 const float* __restrict__ wdwc,
                                                 const float* __restrict__ bdwc,
                                                 bf16_t* __restrict__ fmtok)
{
    const int gm = blockIdx.x;                 // token
    const int c = threadIdx.x;                 // full channel (head*32 + cd)
    const int cd = c & (HD - 1);
    const int b = gm >> 14, n = gm & (NTOK - 1);
    const int h = n >> 7, w = n & 127;
    float a = bdwc[cd];
    #pragma unroll
    for (int dh = -1; dh <= 1; ++dh)
        #pragma unroll
        for (int dw = -1; dw <= 1; ++dw) {
            const int hh = h + dh, ww = w + dw;
            if (hh >= 0 && hh < IMH && ww >= 0 && ww < IMW)
                a += wdwc[cd * 9 + (dh + 1) * 3 + (dw + 1)] *
                     bf2f(vbf[((size_t)(b * NTOK + hh * IMW + ww)) * CHANS + c]);
        }
    fmtok[(size_t)gm * CHANS + c] = f2bf(a);
}

// ---------------------------------------------------------------------------
// 9) out = (q . kv) * z + fm ; WMMA M=64-token tile, N=32 per bh; one wave/block.
//    A rows read straight from global (contiguous 32-half head slice per token).
// ---------------------------------------------------------------------------
__global__ __launch_bounds__(32) void out_kernel(const bf16_t* __restrict__ qbf,
                                                 const bf16_t* __restrict__ kvbf,
                                                 const float* __restrict__ zb,
                                                 const bf16_t* __restrict__ fmtok,
                                                 bf16_t* __restrict__ outtok)
{
    const int bh = blockIdx.y, b = bh >> 3, head = bh & 7;
    const int tile = blockIdx.x * 64;
    const int lane = threadIdx.x, lh = lane & 15;

    v16bf bfrag[2];
    #pragma unroll
    for (int nf = 0; nf < 2; ++nf) {
        const int d = nf * 16 + lh;
        bfrag[nf] = ldB16(kvbf + (size_t)bh * (HD * HD) + d * HD, lane);
    }

    #pragma unroll
    for (int mf = 0; mf < 4; ++mf) {
        const int mrow = tile + mf * 16 + lh;
        const v16bf afrag = ldA16(
            qbf + ((size_t)(b * NTOK + mrow)) * CHANS + head * HD, lane);
        #pragma unroll
        for (int nf = 0; nf < 2; ++nf) {
            v8f acc = {};
            acc = __builtin_amdgcn_wmma_f32_16x16x32_bf16(
                false, afrag, false, bfrag[nf], (short)0, acc, false, false);
            const int cglob = head * HD + nf * 16 + lh;
            #pragma unroll
            for (int r = 0; r < 8; ++r) {
                const int n = tile + mf * 16 + r + ((lane & 16) ? 8 : 0);
                const float zz = zb[(size_t)bh * NTOK + n];
                const size_t idx = ((size_t)(b * NTOK + n)) * CHANS + cglob;
                const float fmv = bf2f(fmtok[idx]);
                outtok[idx] = f2bf(acc[r] * zz + fmv);
            }
        }
    }
}

// ---------------------------------------------------------------------------
// Host orchestration
// ---------------------------------------------------------------------------
extern "C" void kernel_launch(void* const* d_in, const int* in_sizes, int n_in,
                              void* d_out, int out_size, void* d_ws, size_t ws_size,
                              hipStream_t stream)
{
    (void)in_sizes; (void)n_in; (void)out_size;
    const float* x    = (const float*)d_in[0];
    const float* p    = (const float*)d_in[1];
    const float* wq   = (const float*)d_in[2];
    const float* bq   = (const float*)d_in[3];
    const float* wk   = (const float*)d_in[4];
    const float* bk   = (const float*)d_in[5];
    const float* wv   = (const float*)d_in[6];
    const float* bv   = (const float*)d_in[7];
    const float* wdwc = (const float*)d_in[8];
    const float* bdwc = (const float*)d_in[9];
    const float* wx   = (const float*)d_in[10];
    const float* bx   = (const float*)d_in[11];
    const float* wpp  = (const float*)d_in[12];
    const float* bpp  = (const float*)d_in[13];

    float* xout = (float*)d_out;                       // [4,256,128,128]
    float* pout = xout + (size_t)BATCH * CHANS * NTOK; // [4,256,128,128]

    unsigned char* ws = (unsigned char*)d_ws;
    size_t off = 0;
    auto take = [&](size_t sz) -> size_t {
        size_t r = off; off = (off + sz + 255) & ~(size_t)255; return r;
    };
    const size_t SZ_TOKBF = (size_t)BN * CHANS * sizeof(bf16_t);  // 32 MB
    const size_t SZ_TOKF  = (size_t)BN * CHANS * sizeof(float);   // 64 MB
    const size_t SZ_W     = (size_t)CHANS * KCONV * sizeof(bf16_t);

    const size_t o_xbf  = take(SZ_TOKBF);
    const size_t o_pbf  = take(SZ_TOKBF);
    const size_t o_wq   = take(SZ_W);
    const size_t o_wk   = take(SZ_W);
    const size_t o_wv   = take(SZ_W);
    const size_t o_wx   = take(SZ_W);
    const size_t o_wp   = take(SZ_W);
    const size_t o_qf   = take(SZ_TOKF);
    const size_t o_kf   = take(SZ_TOKF);
    const size_t o_vf   = take(SZ_TOKF);
    const size_t o_qbf  = take(SZ_TOKBF);
    const size_t o_kbf  = take(SZ_TOKBF);
    const size_t o_vbf  = take(SZ_TOKBF);
    const size_t o_ksum = take((size_t)BATCH * CHANS * sizeof(float));
    const size_t o_z    = take((size_t)BATCH * NHEADS * NTOK * sizeof(float));
    const size_t o_kv   = take((size_t)BATCH * NHEADS * HD * HD * sizeof(bf16_t));
    // aliases into buffers that are dead after normalize_kernel:
    const size_t o_fmtok  = o_qf;   // 32 MB inside 64 MB qf region
    const size_t o_outtok = o_kf;   // 32 MB inside 64 MB kf region
    if (ws_size < off) return;

    bf16_t* xbf  = (bf16_t*)(ws + o_xbf);
    bf16_t* pbf  = (bf16_t*)(ws + o_pbf);
    bf16_t* wqp  = (bf16_t*)(ws + o_wq);
    bf16_t* wkp  = (bf16_t*)(ws + o_wk);
    bf16_t* wvp  = (bf16_t*)(ws + o_wv);
    bf16_t* wxp  = (bf16_t*)(ws + o_wx);
    bf16_t* wpp2 = (bf16_t*)(ws + o_wp);
    float*  qf   = (float*)(ws + o_qf);
    float*  kf   = (float*)(ws + o_kf);
    float*  vf   = (float*)(ws + o_vf);
    bf16_t* qbf  = (bf16_t*)(ws + o_qbf);
    bf16_t* kbf  = (bf16_t*)(ws + o_kbf);
    bf16_t* vbf  = (bf16_t*)(ws + o_vbf);
    float*  ksum = (float*)(ws + o_ksum);
    float*  zb   = (float*)(ws + o_z);
    bf16_t* kvbf = (bf16_t*)(ws + o_kv);
    bf16_t* fmtok  = (bf16_t*)(ws + o_fmtok);
    bf16_t* outtok = (bf16_t*)(ws + o_outtok);

    // 1) repack weights to bf16 [Cout][K]
    prep_w_kernel<<<CHANS, 256, 0, stream>>>(wq, wqp);
    prep_w_kernel<<<CHANS, 256, 0, stream>>>(wk, wkp);
    prep_w_kernel<<<CHANS, 256, 0, stream>>>(wv, wvp);
    prep_w_kernel<<<CHANS, 256, 0, stream>>>(wx, wxp);
    prep_w_kernel<<<CHANS, 256, 0, stream>>>(wpp, wpp2);

    // 2) NCHW -> token-major bf16
    tok_bf16_kernel<<<BATCH * CHANS, 256, 0, stream>>>(x, xbf);
    tok_bf16_kernel<<<BATCH * CHANS, 256, 0, stream>>>(p, pbf);

    // 3) q/k/v convs (WMMA implicit GEMM, async-LDS pipelined), token-major f32 out
    dim3 cgrid(BN / 64, CHANS / 64);
    conv3x3_wmma_kernel<<<cgrid, 128, 0, stream>>>(xbf, wqp, bq, qf, 0);
    conv3x3_wmma_kernel<<<cgrid, 128, 0, stream>>>(pbf, wkp, bk, kf, 0);
    conv3x3_wmma_kernel<<<cgrid, 128, 0, stream>>>(pbf, wvp, bv, vf, 0);

    // 4) focus + normalize (+ ksum)
    zero_kernel<<<(BATCH * CHANS + 255) / 256, 256, 0, stream>>>(ksum, BATCH * CHANS);
    normalize_kernel<<<BN / 8, 256, 0, stream>>>(qf, kf, vf, qbf, kbf, vbf, ksum);

    // 5) z, kv, fm
    z_kernel<<<(BN * NHEADS) / 256, 256, 0, stream>>>(qbf, ksum, zb);
    kv_kernel<<<BATCH * NHEADS, 128, 0, stream>>>(kbf, vbf, kvbf);
    fm_kernel<<<BN, 256, 0, stream>>>(vbf, wdwc, bdwc, fmtok);

    // 6) out = q.kv * z + fm  (WMMA)
    out_kernel<<<dim3(NTOK / 64, BATCH * NHEADS), 32, 0, stream>>>(
        qbf, kvbf, zb, fmtok, outtok);

    // 7) final convs -> NCHW f32 into d_out
    conv3x3_wmma_kernel<<<cgrid, 128, 0, stream>>>(outtok, wxp, bx, xout, 1);
    conv3x3_wmma_kernel<<<cgrid, 128, 0, stream>>>(fmtok, wpp2, bpp, pout, 1);
}
